// FcNCA_31396210933865
// MI455X (gfx1250) — compile-verified, matrix operand order checked by reading, not run on hardware
//
#include <hip/hip_runtime.h>

typedef __attribute__((ext_vector_type(16))) _Float16 v16h;
typedef __attribute__((ext_vector_type(8)))  float    v8f;
typedef __attribute__((ext_vector_type(4)))  _Float16 h4;
typedef __attribute__((ext_vector_type(4)))  float    f4;

#define BB   16
#define HH   128
#define WWD  128
#define CC   16
#define NPIX (BB*HH*WWD)      // 262144 pixels
#define K1   160              // 144 padded to multiple of 32
#define FIRE 0.5f

__device__ __constant__ int c_dH[8] = {-1, 0, 1,-1, 1,-1, 0, 1};
__device__ __constant__ int c_dW[8] = {-1,-1,-1, 0, 0, 1, 1, 1};

// K position of element e (0..15) of a 16-bit A/B fragment for `lane`,
// per CDNA5 ISA 16-bit A-matrix 16x32 layout (used only by the pack kernel).
__device__ __forceinline__ int kpos(int e, int lane) {
    int v = e >> 1;
    return ((lane & 16) ? 8 : 0) + ((v & 4) ? 16 : 0) + ((v & 3) << 1) + (e & 1);
}

// Fragment-packed LDS tiles: element (k, m) of fragment kt lives at
//   lane = m | ((k&8)<<1) ; e = (k&7) | ((k>>1)&8) ; addr = (kt*32+lane)*16 + e
// so an A fragment is one contiguous 32-byte LDS read per lane.
__device__ __forceinline__ v16h load_afrag(const _Float16* t, int kt, int lane) {
    return *(const v16h*)(t + (size_t)(kt * 32 + lane) * 16);
}

// store hidden value h[N=nt*16+n][M=mrow] into a fragment-packed tile (K = N dim)
__device__ __forceinline__ void store_h(_Float16* t, int nt, int n, int mrow, float v) {
    int kt     = nt >> 1;
    int lane_t = mrow | ((n & 8) << 1);
    int e      = (n & 7) | ((nt & 1) << 3);
    t[(kt * 32 + lane_t) * 16 + e] = (_Float16)v;
}

// ---- pre-pack weights (f32 -> f16, WMMA B-fragment order) --------------
__global__ void pack_weights(const float* __restrict__ W0,
                             const float* __restrict__ W00,
                             const float* __restrict__ W1,
                             _Float16* __restrict__ w0p,
                             _Float16* __restrict__ w00p,
                             _Float16* __restrict__ w1p) {
    int idx = blockIdx.x * blockDim.x + threadIdx.x;
    if (idx < 5 * 16 * 512) {                         // W0: K=160(144), N=256
        int e = idx & 15, lane = (idx >> 4) & 31, r = idx >> 9;
        int nt = r & 15, kt = r >> 4;
        int K = kt * 32 + kpos(e, lane);
        int N = nt * 16 + (lane & 15);
        w0p[idx] = (_Float16)((K < 144) ? W0[N * 144 + K] : 0.0f);
        return;
    }
    idx -= 5 * 16 * 512;
    if (idx < 8 * 8 * 512) {                          // W00: K=256, N=128
        int e = idx & 15, lane = (idx >> 4) & 31, r = idx >> 9;
        int nt = r & 7, kt = r >> 3;
        int K = kt * 32 + kpos(e, lane);
        int N = nt * 16 + (lane & 15);
        w00p[idx] = (_Float16)W00[N * 256 + K];
        return;
    }
    idx -= 8 * 8 * 512;
    if (idx < 4 * 512) {                              // W1: K=128, N=16
        int e = idx & 15, lane = (idx >> 4) & 31, kt = idx >> 9;
        int K = kt * 32 + kpos(e, lane);
        int N = lane & 15;
        w1p[idx] = (_Float16)W1[N * 128 + K];
    }
}

// ---- fused NCA step: perceive + 3-layer MLP (WMMA) + masked update -----
__global__ __launch_bounds__(128)
void nca_step(const float* __restrict__ xin, const float* __restrict__ ustep,
              float* __restrict__ xout,
              const _Float16* __restrict__ w0p, const _Float16* __restrict__ w00p,
              const _Float16* __restrict__ w1p,
              const float* __restrict__ b0, const float* __restrict__ b00) {
    // per-wave: A-tile 5 frags (2560 h) + H-tile 8 frags (4096 h), frag-packed
    __shared__ __align__(64) _Float16 lds[4][2560 + 4096];
    const int lane = threadIdx.x & 31;
    const int wv   = threadIdx.x >> 5;
    _Float16* As = &lds[wv][0];
    _Float16* Hs = &lds[wv][2560];

    const int pixBase = blockIdx.x * 64 + wv * 16;
    const int n  = lane & 15;                  // output column / channel
    const int mb = (lane & 16) ? 8 : 0;        // C/D row base for this half-wave

    // ---- perceive: 16 pixels x 160 feats, float4 loads -> packed f16x4 LDS ----
    for (int i = lane; i < 16 * (K1 / 4); i += 32) {   // 640 groups of 4 feats
        int m  = i / 40;
        int g  = i - m * 40;
        int k0 = g * 4;
        int P  = pixBase + m;
        int b = P >> 14, rem = P & 16383;
        int h = rem >> 7, w = rem & 127;
        f4 v = {0.0f, 0.0f, 0.0f, 0.0f};
        if (k0 < 16) {
            v = *(const f4*)(xin + (size_t)P * 16 + k0);
        } else {
            int q = k0 - 16, ni = q >> 4;
            if (ni < 8) {
                int hh = h + c_dH[ni], ww = w + c_dW[ni];
                if (hh >= 0 && hh < HH && ww >= 0 && ww < WWD)
                    v = *(const f4*)(xin + (size_t)((((b << 7) + hh) << 7) + ww) * 16 + (q & 15));
            }
        }
        h4 hv;
#pragma unroll
        for (int j = 0; j < 4; ++j) hv[j] = (_Float16)v[j];
        int kk = k0 & 31, kt = k0 >> 5;
        int lane_t = m | ((kk & 8) << 1);
        int e0     = (kk & 7) | ((kk >> 1) & 8);      // multiple of 4 -> 8B aligned
        *(h4*)(As + (size_t)(kt * 32 + lane_t) * 16 + e0) = hv;
    }

    // ---- layer 1: [16 x 160] x [160 x 256] ----
    v16h a1[5];
#pragma unroll
    for (int kt = 0; kt < 5; ++kt) a1[kt] = load_afrag(As, kt, lane);

    for (int nt = 0; nt < 16; ++nt) {
        float bias = b0[nt * 16 + n];
        v8f acc;
#pragma unroll
        for (int i = 0; i < 8; ++i) acc[i] = bias;
#pragma unroll
        for (int kt = 0; kt < 5; ++kt) {
            v16h bf = *(const v16h*)(w0p + (size_t)((kt * 16 + nt) * 32 + lane) * 16);
            acc = __builtin_amdgcn_wmma_f32_16x16x32_f16(false, a1[kt], false, bf,
                                                         (short)0, acc, false, false);
        }
#pragma unroll
        for (int r = 0; r < 8; ++r) {
            float hv = acc[r] > 0.0f ? acc[r] : 0.0f;       // ReLU
            store_h(Hs, nt, n, r + mb, hv);
        }
    }

    // ---- layer 2: [16 x 256] x [256 x 128] (h2 overwrites Hs) ----
    v16h a2[8];
#pragma unroll
    for (int kt = 0; kt < 8; ++kt) a2[kt] = load_afrag(Hs, kt, lane);

    for (int nt = 0; nt < 8; ++nt) {
        float bias = b00[nt * 16 + n];
        v8f acc;
#pragma unroll
        for (int i = 0; i < 8; ++i) acc[i] = bias;
#pragma unroll
        for (int kt = 0; kt < 8; ++kt) {
            v16h bf = *(const v16h*)(w00p + (size_t)((kt * 8 + nt) * 32 + lane) * 16);
            acc = __builtin_amdgcn_wmma_f32_16x16x32_f16(false, a2[kt], false, bf,
                                                         (short)0, acc, false, false);
        }
#pragma unroll
        for (int r = 0; r < 8; ++r) {
            float hv = acc[r] > 0.0f ? acc[r] : 0.0f;
            store_h(Hs, nt, n, r + mb, hv);
        }
    }

    // ---- layer 3: [16 x 128] x [128 x 16] -> dx ----
    v16h a3[4];
#pragma unroll
    for (int kt = 0; kt < 4; ++kt) a3[kt] = load_afrag(Hs, kt, lane);

    v8f dx;
#pragma unroll
    for (int i = 0; i < 8; ++i) dx[i] = 0.0f;
#pragma unroll
    for (int kt = 0; kt < 4; ++kt) {
        v16h bf = *(const v16h*)(w1p + (size_t)(kt * 32 + lane) * 16);
        dx = __builtin_amdgcn_wmma_f32_16x16x32_f16(false, a3[kt], false, bf,
                                                    (short)0, dx, false, false);
    }

    // ---- masked update; channels 0..2 pass through ----
#pragma unroll
    for (int r = 0; r < 8; ++r) {
        int P = pixBase + mb + r;
        float xo = xin[(size_t)P * 16 + n];
        float res = xo;
        if (n >= 3) {
            float m = (ustep[P] > FIRE) ? 1.0f : 0.0f;
            res = xo + dx[r] * m;
        }
        xout[(size_t)P * 16 + n] = res;
    }
}

// ---- mean over H,W per (batch, channel) --------------------------------
__global__ void pool_kernel(const float* __restrict__ xf, float* __restrict__ pooled) {
    __shared__ float red[256];
    int b = blockIdx.x;
    int c = threadIdx.x & 15;
    int g = threadIdx.x >> 4;
    float s = 0.0f;
    for (int p = g; p < HH * WWD; p += 16)
        s += xf[((size_t)b * HH * WWD + p) * 16 + c];
    red[threadIdx.x] = s;
    __syncthreads();
    if (g == 0) {
        float t = 0.0f;
#pragma unroll
        for (int i = 0; i < 16; ++i) t += red[i * 16 + c];
        pooled[b * 16 + c] = t * (1.0f / (HH * WWD));
    }
}

// ---- tiny classifier head ----------------------------------------------
__global__ void head_kernel(const float* __restrict__ pooled,
                            const float* __restrict__ W2, const float* __restrict__ b2,
                            const float* __restrict__ W3, const float* __restrict__ b3,
                            float* __restrict__ out) {
    __shared__ float pl[256];
    __shared__ float hid[16 * 128];
    int t = threadIdx.x;
    pl[t] = pooled[t];
    __syncthreads();
    for (int i = t; i < 16 * 128; i += 256) {
        int b = i >> 7, j = i & 127;
        float s = b2[j];
#pragma unroll
        for (int c = 0; c < 16; ++c) s += pl[b * 16 + c] * W2[j * 16 + c];
        hid[i] = s > 0.0f ? s : 0.0f;
    }
    __syncthreads();
    for (int o = t; o < 16 * 11; o += 256) {
        int b = o / 11, j = o - b * 11;
        float s = b3[j];
        for (int i = 0; i < 128; ++i) s += hid[b * 128 + i] * W3[j * 128 + i];
        out[o] = s;
    }
}

extern "C" void kernel_launch(void* const* d_in, const int* in_sizes, int n_in,
                              void* d_out, int out_size, void* d_ws, size_t ws_size,
                              hipStream_t stream) {
    (void)in_sizes; (void)n_in; (void)out_size; (void)ws_size;
    const float* x   = (const float*)d_in[0];
    const float* u   = (const float*)d_in[1];
    const float* W0  = (const float*)d_in[2];
    const float* b0  = (const float*)d_in[3];
    const float* W00 = (const float*)d_in[4];
    const float* b00 = (const float*)d_in[5];
    const float* W1  = (const float*)d_in[6];
    const float* W2  = (const float*)d_in[7];
    const float* b2  = (const float*)d_in[8];
    const float* W3  = (const float*)d_in[9];
    const float* b3  = (const float*)d_in[10];
    float* out = (float*)d_out;

    char* ws = (char*)d_ws;
    _Float16* w0p  = (_Float16*)(ws);                  //  80 KB
    _Float16* w00p = (_Float16*)(ws + 81920);          //  64 KB
    _Float16* w1p  = (_Float16*)(ws + 147456);         //   4 KB
    float* bufA    = (float*)(ws + 151552);            //  16 MB state bounce
    float* pooled  = (float*)(ws + 151552 + (size_t)NPIX * CC * 4);
    float* xf_out  = out + 176;                        // final state lives in d_out

    pack_weights<<<(75776 + 255) / 256, 256, 0, stream>>>(W0, W00, W1, w0p, w00p, w1p);

    // ping-pong so odd steps (incl. step 31) land in xf_out
    const float* cur = x;
    for (int s = 0; s < 32; ++s) {
        float* nxt = (s & 1) ? xf_out : bufA;
        nca_step<<<NPIX / 64, 128, 0, stream>>>(cur, u + (size_t)s * NPIX, nxt,
                                                w0p, w00p, w1p, b0, b00);
        cur = nxt;
    }

    pool_kernel<<<BB, 256, 0, stream>>>(xf_out, pooled);
    head_kernel<<<1, 256, 0, stream>>>(pooled, W2, b2, W3, b3, out);
}
